// WindowAttention_11879879540850
// MI455X (gfx1250) — matrix operand order repaired and checked by reading, not compiled, not run on hardware
//
#include <hip/hip_runtime.h>
#include <hip/hip_bf16.h>

// ---------------------------------------------------------------------------
// Problem constants (from reference)
// ---------------------------------------------------------------------------
#define BB   2
#define NN   4096
#define DIMC 768
#define HH   12
#define DKK  64
#define WW   512
#define NWIN 8              // N / W
#define NBF  128            // feature_size/6
#define RPAD 1056           // padded relative rows (1023 real + pad, covers r<=1039)

typedef __attribute__((ext_vector_type(16))) _Float16 v16h;
typedef __attribute__((ext_vector_type(8)))  float    v8f;
typedef __attribute__((ext_vector_type(4)))  unsigned int u32x4;
typedef __attribute__((ext_vector_type(8)))  int      i32x8;
typedef __attribute__((ext_vector_type(4)))  int      i32x4;

#if __has_builtin(__builtin_amdgcn_tensor_load_to_lds)
#define HAVE_TDM 1
#else
#define HAVE_TDM 0
#endif

// ---------------------------------------------------------------------------
// WMMA helpers (f16 A/B, f32 accumulate, 16x16x32)
// ---------------------------------------------------------------------------
__device__ __forceinline__ v8f wmma_f16(v16h a, v16h b, v8f c) {
    return __builtin_amdgcn_wmma_f32_16x16x32_f16(
        /*neg_a=*/false, a, /*neg_b=*/false, b,
        /*c_mod=*/(short)0, c, /*reuse_a=*/false, /*reuse_b=*/false);
}

// A fragment: 16(M) x 32(K) f16, row-major source [M][K], stride in halves.
// ISA 7.12.2: lane m/m+16; halves 0..7 -> K=hi*8+q, halves 8..15 -> 16+hi*8+q.
__device__ __forceinline__ v16h load_frag_a(const _Float16* __restrict__ base, int stride) {
    int lane = threadIdx.x & 31;
    const _Float16* p = base + (size_t)(lane & 15) * stride + ((lane >> 4) << 3);
    v16h r;
    *((float4*)&r)       = *(const float4*)(p);
    *(((float4*)&r) + 1) = *(const float4*)(p + 16);
    return r;
}

// B fragment: 32(K) x 16(N) f16, source stored [N][K] (K contiguous).
// Lane n/n+16: halves 0..7 -> K=hi*8+q, halves 8..15 -> 16+hi*8+q.
__device__ __forceinline__ v16h load_frag_b(const _Float16* __restrict__ base, int stride) {
    int lane = threadIdx.x & 31;
    const _Float16* p = base + (size_t)(lane & 15) * stride + ((lane >> 4) << 3);
    v16h r;
    *((float4*)&r)       = *(const float4*)(p);
    *(((float4*)&r) + 1) = *(const float4*)(p + 16);
    return r;
}

// ---------------------------------------------------------------------------
// TDM: 2-D tile (nrows x rowlen f16 elements, row stride stride_elts) from
// global to LDS byte offset lds_off. D# per CDNA5 ISA.
// ---------------------------------------------------------------------------
#if HAVE_TDM
__device__ __forceinline__ void tdm_load_2d_f16(unsigned int lds_off, const void* gptr,
                                                unsigned int rowlen_elts,
                                                unsigned int nrows,
                                                unsigned int stride_elts) {
    unsigned long long ga = (unsigned long long)(uintptr_t)gptr;
    u32x4 g0 = { 1u,                                     // count=1, user mode
                 lds_off,                                // lds_addr (bytes)
                 (unsigned int)ga,                       // global_addr[31:0]
                 (unsigned int)((ga >> 32) & 0x01FFFFFFu) | 0x80000000u }; // [56:32] | type=2
    i32x8 g1 = { (int)0x00010000u,                       // data_size=1 (2 bytes), mask=0
                 (int)(rowlen_elts << 16),               // tensor_dim0[15:0] @ bits 63:48
                 (int)(nrows << 16),                     // dim0 hi=0 | tensor_dim1[15:0]
                 (int)(rowlen_elts << 16),               // dim1 hi=0 | tile_dim0
                 (int)nrows,                             // tile_dim1 | tile_dim2=0
                 (int)stride_elts,                       // tensor_dim0_stride lo32
                 0, 0 };                                 // stride hi | dim1_stride=0
    i32x4 gz = { 0, 0, 0, 0 };
#if __clang_major__ >= 23
    i32x8 gz8 = { 0, 0, 0, 0, 0, 0, 0, 0 };
    __builtin_amdgcn_tensor_load_to_lds(g0, g1, gz, gz, gz8, 0);
#else
    __builtin_amdgcn_tensor_load_to_lds(g0, g1, gz, gz, 0);
#endif
}
#endif

// ---------------------------------------------------------------------------
// K1: Enformer positional features -> pos[1024][768] f16 (row 1023 zero pad)
// ---------------------------------------------------------------------------
__global__ void pos_embed_kernel(_Float16* __restrict__ pos) {
    __shared__ float red[NBF];
    const int r = blockIdx.x;       // 0..1023
    const int j = threadIdx.x;      // 0..127
    _Float16* row = pos + (size_t)r * DIMC;
    if (r >= 2 * WW - 1) {          // pad row
        #pragma unroll
        for (int t = 0; t < 6; ++t) row[t * NBF + j] = (_Float16)0.0f;
        return;
    }
    const float dist = (float)(r - (WW - 1));
    const float p = fabsf(dist);
    const float hl = exp2f(3.0f + 6.0f * (float)j / 127.0f);
    const float fe = exp2f(-p / hl);
    const float cw = exp2f((float)(j + 1)) - 1.0f;
    const float fc = (cw > p) ? 1.0f : 0.0f;
    const float mean = 4.0f + (508.0f / 127.0f) * (float)j;
    const float conc = 0.25f * mean * mean;   // (mean/2)^2, stddev=2
    const float rate = 0.25f * mean;          // mean/4
    const float xl   = (p > 0.0f) ? (conc - 1.0f) * logf(p) : 0.0f;
    const float lp   = xl - rate * p - (lgammaf(conc) - conc * logf(rate));
    const float prob = expf(lp) + 1e-8f;
    red[j] = prob;
    __syncthreads();
    for (int s = NBF / 2; s > 0; s >>= 1) {
        if (j < s) red[j] = fmaxf(red[j], red[j + s]);
        __syncthreads();
    }
    const float fg = prob / red[0];
    const float sg = (dist > 0.0f) ? 1.0f : ((dist < 0.0f) ? -1.0f : 0.0f);
    row[0 * NBF + j] = (_Float16)fe;
    row[1 * NBF + j] = (_Float16)fc;
    row[2 * NBF + j] = (_Float16)fg;
    row[3 * NBF + j] = (_Float16)(sg * fe);
    row[4 * NBF + j] = (_Float16)(sg * fc);
    row[5 * NBF + j] = (_Float16)(sg * fg);
}

// Zero pad rows 1024..1055 of relk[h][RPAD][64]
__global__ void relk_zero_pad(_Float16* __restrict__ relk) {
    int i = blockIdx.x * 256 + threadIdx.x;
    const int total = HH * (RPAD - 1024) * DKK;
    if (i < total) {
        int h = i / ((RPAD - 1024) * DKK);
        int rem = i % ((RPAD - 1024) * DKK);
        relk[((size_t)h * RPAD + 1024) * DKK + rem] = (_Float16)0.0f;
    }
}

// ---------------------------------------------------------------------------
// K2: rel_k = pos @ w_rel -> relk[h][r][d] f16
// ---------------------------------------------------------------------------
__global__ __launch_bounds__(256) void relk_gemm_kernel(
    const _Float16* __restrict__ pos, const float* __restrict__ w_rel,
    _Float16* __restrict__ relk) {
    __shared__ _Float16 Bs[96][40];
    const int tid = threadIdx.x, wave = tid >> 5;
    const int row0 = blockIdx.x * 128, c0 = blockIdx.y * 96;
    v8f acc[6] = {};
    for (int k0 = 0; k0 < DIMC; k0 += 32) {
        __syncthreads();
        for (int i = tid; i < 96 * 32; i += 256) {
            int k = i / 96, c = i % 96;
            Bs[c][k] = (_Float16)w_rel[(size_t)(k0 + k) * DIMC + c0 + c];
        }
        if (k0 + 32 < DIMC) __builtin_prefetch(w_rel + (size_t)(k0 + 32) * DIMC + c0, 0, 1);
        __syncthreads();
        v16h a = load_frag_a(pos + (size_t)(row0 + wave * 16) * DIMC + k0, DIMC);
        #pragma unroll
        for (int ct = 0; ct < 6; ++ct) {
            v16h bf = load_frag_b(&Bs[ct * 16][0], 40);
            acc[ct] = wmma_f16(a, bf, acc[ct]);
        }
    }
    const int lane = tid & 31, hi = lane >> 4, ncol = lane & 15;
    #pragma unroll
    for (int ct = 0; ct < 6; ++ct)
        #pragma unroll
        for (int g = 0; g < 8; ++g) {
            int gr = row0 + wave * 16 + g + hi * 8;
            int gc = c0 + ct * 16 + ncol;             // h*64 + d
            relk[((size_t)(gc >> 6) * RPAD + gr) * DKK + (gc & 63)] = (_Float16)acc[ct][g];
        }
}

// ---------------------------------------------------------------------------
// K3: fused QKV projection; scatter-stores qc=q+cb, qp=q+pb, k, vT (f16)
// ---------------------------------------------------------------------------
__global__ __launch_bounds__(256) void qkv_gemm_kernel(
    const float* __restrict__ x, const float* __restrict__ wq,
    const float* __restrict__ cb, const float* __restrict__ pb,
    _Float16* __restrict__ qc, _Float16* __restrict__ qp,
    _Float16* __restrict__ kmat, _Float16* __restrict__ vT) {
    __shared__ _Float16 As[128][40];
    __shared__ _Float16 Bs[96][40];
    const int tid = threadIdx.x, wave = tid >> 5;
    const int row0 = blockIdx.x * 128, c0 = blockIdx.y * 96;
    v8f acc[6] = {};
    for (int k0 = 0; k0 < DIMC; k0 += 32) {
        __syncthreads();
        for (int i = tid; i < 128 * 32; i += 256) {
            int r = i >> 5, k = i & 31;
            As[r][k] = (_Float16)x[(size_t)(row0 + r) * DIMC + k0 + k];
        }
        for (int i = tid; i < 96 * 32; i += 256) {
            int k = i / 96, c = i % 96;
            Bs[c][k] = (_Float16)wq[(size_t)(k0 + k) * (3 * HH * DKK) + c0 + c];
        }
        if (k0 + 32 < DIMC) __builtin_prefetch(wq + (size_t)(k0 + 32) * (3 * HH * DKK) + c0, 0, 1);
        __syncthreads();
        v16h a = load_frag_a(&As[wave * 16][0], 40);
        #pragma unroll
        for (int ct = 0; ct < 6; ++ct) {
            v16h bf = load_frag_b(&Bs[ct * 16][0], 40);
            acc[ct] = wmma_f16(a, bf, acc[ct]);
        }
    }
    const int lane = tid & 31, hi = lane >> 4, ncol = lane & 15;
    #pragma unroll
    for (int ct = 0; ct < 6; ++ct)
        #pragma unroll
        for (int g = 0; g < 8; ++g) {
            int gr = row0 + wave * 16 + g + hi * 8;     // b*4096+n
            int gc = c0 + ct * 16 + ncol;               // (h*64+d)*3 + t
            float v = acc[ct][g];
            int t = gc % 3, hd = gc / 3;
            int h = hd >> 6, d = hd & 63;
            int batch = gr >> 12, n = gr & (NN - 1);
            size_t qi = ((size_t)(batch * HH + h) * NN + n) * DKK + d;
            if (t == 0) {
                qc[qi] = (_Float16)(v + cb[hd]);
                qp[qi] = (_Float16)(v + pb[hd]);
            } else if (t == 1) {
                kmat[qi] = (_Float16)v;
            } else {
                vT[((size_t)(batch * HH + h) * DKK + d) * NN + n] = (_Float16)v;
            }
        }
}

// ---------------------------------------------------------------------------
// K4: fused windowed attention. One block = 32 query rows of one (b,h,window).
// LDS: Rb[32][560] f32 | S[32][512] f32 | KV[64KB] (K tile, then V tile via
// TDM); P[32][512] f16 aliases Rb. Dynamic LDS 198 KB (< 320 KB WGP limit).
// ---------------------------------------------------------------------------
#define RBW 560
#define RB_BYTES (32 * RBW * 4)          // 71680
#define S_BYTES  (32 * WW * 4)           // 65536
#define KV_BYTES (WW * DKK * 2)          // 65536
#define ATTN_LDS_BYTES (RB_BYTES + S_BYTES + KV_BYTES)

__global__ __launch_bounds__(256) void attn_kernel(
    const _Float16* __restrict__ qc, const _Float16* __restrict__ qp,
    const _Float16* __restrict__ kmat, const _Float16* __restrict__ vT,
    const _Float16* __restrict__ relk, _Float16* __restrict__ o) {
    extern __shared__ char smem[];
    float*    Rb = (float*)smem;                       // 32 x 560 f32
    float*    S  = (float*)(smem + RB_BYTES);          // 32 x 512 f32
    _Float16* KV = (_Float16*)(smem + RB_BYTES + S_BYTES);
    _Float16* P  = (_Float16*)smem;                    // aliases Rb (dead after ph2)

    const int tid  = threadIdx.x, wave = tid >> 5;
    const int lane = tid & 31, hi = lane >> 4, ncol = lane & 15;
    const int idx  = blockIdx.x;
    const int i0   = (idx & 15) * 32;
    const int wwin = (idx >> 4) & 7;
    const int head = (idx >> 7) % HH;
    const int batch = idx / (16 * 8 * HH);
    const int bh = batch * HH + head;
    const size_t qbase = ((size_t)bh * NN + wwin * WW + i0) * DKK;
    const size_t kbase = ((size_t)bh * NN + wwin * WW) * DKK;
    const int r0 = 480 - i0;                  // band base: gather idx = j-(i-i0)+31

    // Kick off async K-tile DMA (512 rows x 64 f16, stride 64) — overlaps phase 1.
#if HAVE_TDM
    const unsigned int kv_lds = __builtin_amdgcn_groupstaticsize() + RB_BYTES + S_BYTES;
    if (wave == 0)
        tdm_load_2d_f16(kv_lds, kmat + kbase, DKK, WW, DKK);
#endif

    // ---- phase 1: rel band  Rb = (q+pb) @ rel_k^T  (2 x 35 tiles) ----
    for (int t = wave; t < 2 * 35; t += 8) {
        int mt = t / 35, ct = t % 35;
        v8f acc = {};
        #pragma unroll
        for (int k0 = 0; k0 < DKK; k0 += 32) {
            v16h a  = load_frag_a(qp + qbase + (size_t)mt * 16 * DKK + k0, DKK);
            v16h bf = load_frag_b(relk + ((size_t)head * RPAD + r0 + ct * 16) * DKK + k0, DKK);
            acc = wmma_f16(a, bf, acc);
        }
        #pragma unroll
        for (int g = 0; g < 8; ++g)
            Rb[(mt * 16 + g + hi * 8) * RBW + ct * 16 + ncol] = acc[g];
    }
#if HAVE_TDM
    if (wave == 0) __builtin_amdgcn_s_wait_tensorcnt(0);
#else
    for (int i = tid; i < WW * DKK / 8; i += 256)
        *(((float4*)KV) + i) = *(((const float4*)(kmat + kbase)) + i);
#endif
    __syncthreads();

    // ---- phase 2: scores  S = (q+cb)@K_lds^T * 1/8 + shifted rel ----
    for (int t = wave; t < 2 * 32; t += 8) {
        int mt = t >> 5, jt = t & 31;
        v8f acc = {};
        #pragma unroll
        for (int k0 = 0; k0 < DKK; k0 += 32) {
            v16h a  = load_frag_a(qc + qbase + (size_t)mt * 16 * DKK + k0, DKK);
            v16h bf = load_frag_b(KV + (size_t)jt * 16 * DKK + k0, DKK);
            acc = wmma_f16(a, bf, acc);
        }
        #pragma unroll
        for (int g = 0; g < 8; ++g) {
            int il = mt * 16 + g + hi * 8;
            int j  = jt * 16 + ncol;
            S[il * WW + j] = acc[g] * 0.125f + Rb[il * RBW + (j - il + 31)];
        }
    }
    __syncthreads();

    // Async V-tile DMA into same LDS region (64 rows x 512 f16, stride NN) —
    // overlaps the softmax phase below.
#if HAVE_TDM
    if (wave == 0)
        tdm_load_2d_f16(kv_lds, vT + (size_t)bh * DKK * NN + wwin * WW, WW, DKK, NN);
#endif

    // ---- phase 3: softmax rows, write f16 P (aliases Rb region) ----
    {
        int row = tid >> 3, sub = tid & 7;     // 8 threads per row
        float m = -3.4e38f;
        for (int j = sub; j < WW; j += 8) m = fmaxf(m, S[row * WW + j]);
        m = fmaxf(m, __shfl_xor(m, 1));
        m = fmaxf(m, __shfl_xor(m, 2));
        m = fmaxf(m, __shfl_xor(m, 4));
        float s = 0.0f;
        for (int j = sub; j < WW; j += 8) {
            float e = expf(S[row * WW + j] - m);
            S[row * WW + j] = e;
            s += e;
        }
        s += __shfl_xor(s, 1);
        s += __shfl_xor(s, 2);
        s += __shfl_xor(s, 4);
        float inv = 1.0f / s;
        for (int j = sub; j < WW; j += 8)
            P[row * WW + j] = (_Float16)(S[row * WW + j] * inv);
    }
#if HAVE_TDM
    if (wave == 0) __builtin_amdgcn_s_wait_tensorcnt(0);
#else
    for (int i = tid; i < WW * DKK / 8; i += 256) {
        int d = i / (WW / 8), c8 = i % (WW / 8);
        *(((float4*)(KV + (size_t)d * WW)) + c8) =
            *(((const float4*)(vT + ((size_t)bh * DKK + d) * NN + wwin * WW)) + c8);
    }
#endif
    __syncthreads();

    // ---- phase 4: O = P @ V_lds  (one 16x16 tile per wave) ----
    {
        int mt = wave >> 2, nt = wave & 3;
        v8f acc = {};
        #pragma unroll 4
        for (int s0 = 0; s0 < WW; s0 += 32) {
            v16h a  = load_frag_a(P + (size_t)mt * 16 * WW + s0, WW);
            v16h bf = load_frag_b(KV + (size_t)nt * 16 * WW + s0, WW);
            acc = wmma_f16(a, bf, acc);
        }
        int ng = wwin * WW + i0 + mt * 16;
        #pragma unroll
        for (int g = 0; g < 8; ++g)
            o[((size_t)batch * NN + ng + g + hi * 8) * DIMC + head * DKK + nt * 16 + ncol] =
                (_Float16)acc[g];
    }
}

// ---------------------------------------------------------------------------
// K5: output projection  out = o @ w_out + b_out  (fp32 out)
// ---------------------------------------------------------------------------
__global__ __launch_bounds__(256) void out_gemm_kernel(
    const _Float16* __restrict__ o, const float* __restrict__ w_out,
    const float* __restrict__ b_out, float* __restrict__ out) {
    __shared__ _Float16 Bs[96][40];
    const int tid = threadIdx.x, wave = tid >> 5;
    const int row0 = blockIdx.x * 128, c0 = blockIdx.y * 96;
    v8f acc[6] = {};
    for (int k0 = 0; k0 < DIMC; k0 += 32) {
        __syncthreads();
        for (int i = tid; i < 96 * 32; i += 256) {
            int k = i / 96, c = i % 96;
            Bs[c][k] = (_Float16)w_out[(size_t)(k0 + k) * DIMC + c0 + c];
        }
        if (k0 + 32 < DIMC) __builtin_prefetch(w_out + (size_t)(k0 + 32) * DIMC + c0, 0, 1);
        __syncthreads();
        v16h a = load_frag_a(o + (size_t)(row0 + wave * 16) * DIMC + k0, DIMC);
        #pragma unroll
        for (int ct = 0; ct < 6; ++ct) {
            v16h bf = load_frag_b(&Bs[ct * 16][0], 40);
            acc[ct] = wmma_f16(a, bf, acc[ct]);
        }
    }
    const int lane = tid & 31, hi = lane >> 4, ncol = lane & 15;
    #pragma unroll
    for (int ct = 0; ct < 6; ++ct)
        #pragma unroll
        for (int g = 0; g < 8; ++g) {
            int gr = row0 + wave * 16 + g + hi * 8;
            int gc = c0 + ct * 16 + ncol;
            out[(size_t)gr * DIMC + gc] = acc[ct][g] + b_out[gc];
        }
}

// ---------------------------------------------------------------------------
// Launch
// ---------------------------------------------------------------------------
extern "C" void kernel_launch(void* const* d_in, const int* in_sizes, int n_in,
                              void* d_out, int out_size, void* d_ws, size_t ws_size,
                              hipStream_t stream) {
    (void)in_sizes; (void)n_in; (void)out_size; (void)ws_size;
    const float* x     = (const float*)d_in[0];
    const float* w_qkv = (const float*)d_in[1];
    const float* w_out = (const float*)d_in[2];
    const float* b_out = (const float*)d_in[3];
    const float* w_rel = (const float*)d_in[4];
    const float* cb    = (const float*)d_in[5];
    const float* pb    = (const float*)d_in[6];

    char* ws = (char*)d_ws;
    const size_t POS_BYTES  = (size_t)1024 * DIMC * 2;
    const size_t RELK_BYTES = (size_t)HH * RPAD * DKK * 2;
    const size_t QLIKE      = (size_t)BB * HH * NN * DKK * 2;
    const size_t O_BYTES    = (size_t)BB * NN * DIMC * 2;

    size_t off = 0;
    _Float16* pos  = (_Float16*)(ws + off); off += POS_BYTES;
    _Float16* relk = (_Float16*)(ws + off); off += RELK_BYTES;
    _Float16* qc   = (_Float16*)(ws + off); off += QLIKE;
    _Float16* qp   = (_Float16*)(ws + off); off += QLIKE;
    _Float16* kmat = (_Float16*)(ws + off); off += QLIKE;
    _Float16* vT   = (_Float16*)(ws + off); off += QLIKE;
    _Float16* oatt = (_Float16*)(ws + off); off += O_BYTES;
    (void)off;  // ~66.1 MB total workspace

    pos_embed_kernel<<<1024, NBF, 0, stream>>>(pos);
    relk_zero_pad<<<(HH * (RPAD - 1024) * DKK + 255) / 256, 256, 0, stream>>>(relk);
    relk_gemm_kernel<<<dim3(8, 8), 256, 0, stream>>>(pos, w_rel, relk);
    qkv_gemm_kernel<<<dim3(64, 24), 256, 0, stream>>>(x, w_qkv, cb, pb, qc, qp, kmat, vT);
    attn_kernel<<<BB * HH * NWIN * 16, 256, ATTN_LDS_BYTES, stream>>>(qc, qp, kmat, vT, relk, oatt);
    out_gemm_kernel<<<dim3(64, 8), 256, 0, stream>>>(oatt, w_out, b_out, (float*)d_out);
}
